// MoIETransformerBlock_11596411699896
// MI455X (gfx1250) — matrix operand order repaired and checked by reading, not compiled, not run on hardware
//
#include <hip/hip_runtime.h>
#include <hip/hip_bf16.h>
#include <math.h>

// ---------------------------------------------------------------------------
// MoIE transformer block for MI455X (gfx1250), wave32 + WMMA f16->f32.
// f16 operands; async global->LDS staging (ASYNCcnt), double buffered;
// 128x128 block tile, 32x64 wave tile (8 WMMA per wave per K-tile).
// ---------------------------------------------------------------------------

typedef __attribute__((ext_vector_type(16))) _Float16 v16h;
typedef __attribute__((ext_vector_type(8)))  _Float16 v8h;
typedef __attribute__((ext_vector_type(4)))  _Float16 v4h;
typedef __attribute__((ext_vector_type(8)))  float    v8f;
typedef __attribute__((ext_vector_type(4)))  int      v4i;

#define TB   2
#define TS   1024
#define TH   2048
#define TNT  (TB * TS)           // tokens = 2048
#define RGC  1.0f

#define BM   128
#define BN   128
#define BK   32
#define LDSK 48                  // padded LDS row stride in halves (96B)

#if defined(__AMDGCN__) && \
    __has_builtin(__builtin_amdgcn_global_load_async_to_lds_b128) && \
    __has_builtin(__builtin_amdgcn_s_wait_asynccnt)
#define USE_ASYNC 1
#else
#define USE_ASYNC 0
#endif

#if USE_ASYNC
// param types per hipcc diagnostic: (AS1 v4i*, AS3 v4i*, imm offset, imm cpol)
#define ASYNC_B128(gp, lp) \
    __builtin_amdgcn_global_load_async_to_lds_b128( \
        (__attribute__((address_space(1))) v4i*)(gp), \
        (__attribute__((address_space(3))) v4i*)(lp), 0, 0)
#endif

// ---------------------------------------------------------------------------
// GEMM: C[M,N] = alpha * A[M,K] x op(B) + bias   (f16 in, f32 out, f16 copy)
// TRANS_B=true : B stored [N,K] row-major (A @ B^T, the x@W.T case)
// TRANS_B=false: B stored [K,N] row-major (A @ B, the probs@V case)
// Block 128x128, 8 waves, wave tile 32 rows x 64 cols (2x4 WMMA subtiles).
// ---------------------------------------------------------------------------
template <bool TRANS_B>
__global__ __launch_bounds__(256, 1)
void gemm_wmma(const _Float16* __restrict__ Ag, const _Float16* __restrict__ Bg,
               const float* __restrict__ bias, float* __restrict__ Cg,
               _Float16* __restrict__ C16g,
               int M, int N, int K, long sA, long sB, long sC, float alpha)
{
    const int tid  = threadIdx.x;
    const int z    = blockIdx.z;
    const _Float16* A = Ag + (long)z * sA;
    const _Float16* B = Bg + (long)z * sB;

    const int n0 = blockIdx.x * BN;
    const int m0 = blockIdx.y * BM;

    const int wave = tid >> 5;
    const int lane = tid & 31;
    const int wm   = wave >> 1;   // 0..3 -> 32-row band
    const int wn   = wave & 1;    // 0..1 -> 64-col band
    const int lh   = lane & 15;
    const int hi   = lane >> 4;

    const int nt = K / BK;
    v8f acc[2][4] = {};

    // fragment builder (ISA VGPR layout): A lanes0-15 K{0-7,16-23}, lanes16-31 K{8-15,24-31}
    auto compute = [&](const _Float16* as, const _Float16* bs) {
        v16h afr[2], bfr[4];
        #pragma unroll
        for (int mi = 0; mi < 2; ++mi) {
            const int m  = wm * 32 + mi * 16 + lh;
            const int kb = hi * 8;
            v8h lo = *(const v8h*)&as[m * LDSK + kb];
            v8h hp = *(const v8h*)&as[m * LDSK + kb + 16];
            afr[mi] = __builtin_shufflevector(lo, hp, 0,1,2,3,4,5,6,7,8,9,10,11,12,13,14,15);
        }
        #pragma unroll
        for (int ni = 0; ni < 4; ++ni) {
            const int n  = wn * 64 + ni * 16 + lh;
            const int kb = hi * 8;
            v8h lo = *(const v8h*)&bs[n * LDSK + kb];
            v8h hp = *(const v8h*)&bs[n * LDSK + kb + 16];
            bfr[ni] = __builtin_shufflevector(lo, hp, 0,1,2,3,4,5,6,7,8,9,10,11,12,13,14,15);
        }
        #pragma unroll
        for (int mi = 0; mi < 2; ++mi)
            #pragma unroll
            for (int ni = 0; ni < 4; ++ni)
                acc[mi][ni] = __builtin_amdgcn_wmma_f32_16x16x32_f16(
                    false, afr[mi], false, bfr[ni], (short)0, acc[mi][ni], false, false);
    };

    constexpr bool ASYNC_TN = (USE_ASYNC != 0) && TRANS_B;

    if constexpr (ASYNC_TN) {
#if USE_ASYNC
        // -------- async double-buffered staging: global -> LDS, no VGPR hop
        __shared__ __align__(16) _Float16 Ad[2][BM * LDSK];
        __shared__ __align__(16) _Float16 Bd[2][BN * LDSK];
        const int srow = tid >> 2;     // rows 0..63 (+64 on second chunk)
        const int sck  = tid & 3;      // 8-half chunk within 32-half row
        auto issue = [&](int k0, int bb) {
            #pragma unroll
            for (int i = 0; i < 2; ++i) {
                const int row = srow + i * 64;
                ASYNC_B128(A + (long)(m0 + row) * K + k0 + sck * 8,
                           &Ad[bb][row * LDSK + sck * 8]);
            }
            #pragma unroll
            for (int i = 0; i < 2; ++i) {
                const int row = srow + i * 64;
                ASYNC_B128(B + (long)(n0 + row) * K + k0 + sck * 8,
                           &Bd[bb][row * LDSK + sck * 8]);
            }
        };
        issue(0, 0);
        for (int t = 0; t + 1 < nt; ++t) {          // steady state: branch-free waits
            issue((t + 1) * BK, (t + 1) & 1);
            __builtin_amdgcn_s_wait_asynccnt(4);    // tile t landed (in-order)
            __syncthreads();
            compute(Ad[t & 1], Bd[t & 1]);
            __syncthreads();
        }
        __builtin_amdgcn_s_wait_asynccnt(0);        // last tile
        __syncthreads();
        compute(Ad[(nt - 1) & 1], Bd[(nt - 1) & 1]);
#endif
    } else {
        // -------- sync path: batched b128 loads, register-pipelined
        __shared__ __align__(16) _Float16 As[BM * LDSK];
        __shared__ __align__(16) _Float16 Bs[BN * LDSK];
        const int srow = tid >> 2, sck = tid & 3;
        const int kr = tid >> 4, nc = tid & 15;   // NN: 32 k-rows x 16 chunks
        v8h ra[2], rb[2];
        auto loadT = [&](int k0) {
            #pragma unroll
            for (int i = 0; i < 2; ++i)
                ra[i] = *(const v8h*)(A + (long)(m0 + srow + i * 64) * K + k0 + sck * 8);
            if constexpr (TRANS_B) {
                #pragma unroll
                for (int i = 0; i < 2; ++i)
                    rb[i] = *(const v8h*)(B + (long)(n0 + srow + i * 64) * K + k0 + sck * 8);
            } else {
                #pragma unroll
                for (int i = 0; i < 2; ++i)
                    rb[i] = *(const v8h*)(B + (long)(k0 + kr + i * 16) * N + n0 + nc * 8);
            }
        };
        loadT(0);
        for (int t = 0; t < nt; ++t) {
            __syncthreads();
            #pragma unroll
            for (int i = 0; i < 2; ++i)
                *(v8h*)&As[(srow + i * 64) * LDSK + sck * 8] = ra[i];
            if constexpr (TRANS_B) {
                #pragma unroll
                for (int i = 0; i < 2; ++i)
                    *(v8h*)&Bs[(srow + i * 64) * LDSK + sck * 8] = rb[i];
            } else {
                #pragma unroll
                for (int i = 0; i < 2; ++i)
                    #pragma unroll
                    for (int j = 0; j < 8; ++j)
                        Bs[(nc * 8 + j) * LDSK + kr + i * 16] = rb[i][j];  // transpose
            }
            __syncthreads();
            if (t + 1 < nt) loadT((t + 1) * BK);           // overlap with WMMA
            compute(As, Bs);
        }
    }

    // -------- epilogue (C/D layout: lanes0-15 rows r, lanes16-31 rows 8+r)
    float*      C   = Cg  + (long)z * sC;
    _Float16*   C16 = C16g ? C16g + (long)z * sC : nullptr;
    #pragma unroll
    for (int mi = 0; mi < 2; ++mi) {
        #pragma unroll
        for (int ni = 0; ni < 4; ++ni) {
            const int col     = n0 + wn * 64 + ni * 16 + lh;
            const float bv    = bias ? bias[col] : 0.0f;
            const int rowbase = m0 + wm * 32 + mi * 16 + hi * 8;
            #pragma unroll
            for (int r = 0; r < 8; ++r) {
                const float v = acc[mi][ni][r] * alpha + bv;
                C[(long)(rowbase + r) * N + col] = v;
                if (C16) C16[(long)(rowbase + r) * N + col] = (_Float16)v;
            }
        }
    }
}

// ---------------------------------------------------------------------------
// Block reductions (256 threads)
// ---------------------------------------------------------------------------
__device__ __forceinline__ float block_sum(float v, int tid) {
    __shared__ float red[256];
    __syncthreads();
    red[tid] = v; __syncthreads();
    for (int s = 128; s > 0; s >>= 1) { if (tid < s) red[tid] += red[tid + s]; __syncthreads(); }
    const float r = red[0]; __syncthreads();
    return r;
}
__device__ __forceinline__ float block_max(float v, int tid) {
    __shared__ float red[256];
    __syncthreads();
    red[tid] = v; __syncthreads();
    for (int s = 128; s > 0; s >>= 1) { if (tid < s) red[tid] = fmaxf(red[tid], red[tid + s]); __syncthreads(); }
    const float r = red[0]; __syncthreads();
    return r;
}

// ---------------------------------------------------------------------------
// f32 -> f16 conversion (4 elems/thread)
// ---------------------------------------------------------------------------
__global__ __launch_bounds__(256)
void cvt16_kernel(const float* __restrict__ a, _Float16* __restrict__ o, long n)
{
    const long i = ((long)blockIdx.x * 256 + threadIdx.x) * 4;
    if (i >= n) return;
    const float4 f = *(const float4*)(a + i);
    v4h h; h[0] = (_Float16)f.x; h[1] = (_Float16)f.y; h[2] = (_Float16)f.z; h[3] = (_Float16)f.w;
    *(v4h*)(o + i) = h;
}

// ---------------------------------------------------------------------------
// LayerNorm: res = LN(in)*g + b (+ add); optional f32 and f16 outputs.
// ---------------------------------------------------------------------------
__global__ __launch_bounds__(256)
void ln_kernel(const float* __restrict__ in, const float* __restrict__ g,
               const float* __restrict__ b, const float* __restrict__ add,
               float* __restrict__ out, _Float16* __restrict__ out16)
{
    const int tid = threadIdx.x;
    const long base = (long)blockIdx.x * TH;
    float xl[8], s = 0.f, sq = 0.f;
    #pragma unroll
    for (int j = 0; j < 8; ++j) {
        const float v = in[base + tid + j * 256];
        xl[j] = v; s += v; sq += v * v;
    }
    const float mean = block_sum(s, tid) * (1.0f / TH);
    const float var  = block_sum(sq, tid) * (1.0f / TH) - mean * mean;
    const float inv  = rsqrtf(var + 1e-5f);
    #pragma unroll
    for (int j = 0; j < 8; ++j) {
        const int i = tid + j * 256;
        float o = (xl[j] - mean) * inv * g[i] + b[i];
        if (add) o += add[base + i];
        if (out)   out[base + i]   = o;
        if (out16) out16[base + i] = (_Float16)o;
    }
}

// ---------------------------------------------------------------------------
// Routing: c = resid+comp; cost = mas(gate); logits = (match-cost)*RG;
//          rw = mas(logits); masked = c*rw.
// ---------------------------------------------------------------------------
__global__ __launch_bounds__(256)
void route_kernel(const float* __restrict__ comp, const float* __restrict__ match,
                  const float* __restrict__ gate, const float* __restrict__ resid,
                  float* __restrict__ masked, float* __restrict__ raw,
                  float* __restrict__ logits)
{
    const int tid = threadIdx.x;
    const long base = (long)blockIdx.x * TH;
    float gl[8], mx = 0.f;
    #pragma unroll
    for (int j = 0; j < 8; ++j) {
        const float v = gate[base + tid + j * 256];
        gl[j] = v; mx = fmaxf(mx, fabsf(v));
    }
    const float gden = block_max(mx, tid) + 1e-9f;
    float ll[8], mx2 = 0.f;
    #pragma unroll
    for (int j = 0; j < 8; ++j) {
        const int i = tid + j * 256;
        const float cost = fmaxf(gl[j] / gden, 0.0f);
        const float l = (match[base + i] - cost) * RGC;
        ll[j] = l; mx2 = fmaxf(mx2, fabsf(l));
    }
    const float lden = block_max(mx2, tid) + 1e-9f;
    #pragma unroll
    for (int j = 0; j < 8; ++j) {
        const int i = tid + j * 256;
        const float rw = fmaxf(ll[j] / lden, 0.0f);
        const float c  = resid[base + i] + comp[base + i];
        masked[base + i] = c * rw;
        raw[base + i]    = rw;
        logits[base + i] = ll[j];
    }
}

// ---------------------------------------------------------------------------
// RoPE: out = x*cos + rot_half(x)*sin ; optional f16 copy
// ---------------------------------------------------------------------------
__global__ __launch_bounds__(256)
void rope_kernel(const float* __restrict__ x, const float* __restrict__ cosb,
                 const float* __restrict__ sinb, float* __restrict__ out,
                 _Float16* __restrict__ out16, long n)
{
    const long idx = (long)blockIdx.x * 256 + threadIdx.x;
    if (idx >= n) return;
    const long bs = idx / TH;
    const int  i  = (int)(idx - bs * TH);
    const int  s  = (int)(bs % TS);
    const int  h  = TH / 2;
    const float rv = (i < h) ? -x[bs * TH + i + h] : x[bs * TH + i - h];
    const float o  = x[idx] * cosb[(long)s * TH + i] + rv * sinb[(long)s * TH + i];
    out[idx] = o;
    if (out16) out16[idx] = (_Float16)o;
}

// ---------------------------------------------------------------------------
// Causal softmax: reads f32 scores[B*S, S], writes f16 probs.
// ---------------------------------------------------------------------------
__global__ __launch_bounds__(256)
void softmax_kernel(const float* __restrict__ sc, _Float16* __restrict__ p16)
{
    const int tid  = threadIdx.x;
    const int row  = blockIdx.x;
    const int qpos = row % TS;
    const float* x = sc + (long)row * TS;
    _Float16*   p  = p16 + (long)row * TS;
    float xl[4], mx = -3.0e38f;
    #pragma unroll
    for (int j = 0; j < 4; ++j) {
        const int i = tid + j * 256;
        const float v = (i <= qpos) ? x[i] : -3.0e38f;
        xl[j] = v; mx = fmaxf(mx, v);
    }
    const float rmax = block_max(mx, tid);
    float s = 0.f;
    #pragma unroll
    for (int j = 0; j < 4; ++j) {
        const int i = tid + j * 256;
        xl[j] = (i <= qpos) ? __expf(xl[j] - rmax) : 0.0f;
        s += xl[j];
    }
    const float rinv = 1.0f / block_sum(s, tid);
    #pragma unroll
    for (int j = 0; j < 4; ++j) p[tid + j * 256] = (_Float16)(xl[j] * rinv);
}

// ---------------------------------------------------------------------------
// Elementwise helpers
// ---------------------------------------------------------------------------
__global__ __launch_bounds__(256)
void add_kernel(const float* __restrict__ a, const float* __restrict__ b,
                float* __restrict__ o, long n)
{
    const long i = (long)blockIdx.x * 256 + threadIdx.x;
    if (i < n) o[i] = a[i] + b[i];
}
__global__ __launch_bounds__(256)
void relu_kernel(const float* __restrict__ a, float* __restrict__ o,
                 _Float16* __restrict__ o16, long n)
{
    const long i = (long)blockIdx.x * 256 + threadIdx.x;
    if (i < n) { const float v = fmaxf(a[i], 0.0f); o[i] = v; o16[i] = (_Float16)v; }
}
__global__ __launch_bounds__(256)
void copy_kernel(const float* __restrict__ a, float* __restrict__ o, long n)
{
    const long i = (long)blockIdx.x * 256 + threadIdx.x;
    if (i < n) o[i] = a[i];
}

// ---------------------------------------------------------------------------
// Orchestration
// ---------------------------------------------------------------------------
extern "C" void kernel_launch(void* const* d_in, const int* in_sizes, int n_in,
                              void* d_out, int out_size, void* d_ws, size_t ws_size,
                              hipStream_t stream)
{
    const float* x     = (const float*)d_in[0];
    const float* cosb  = (const float*)d_in[1];
    const float* sinb  = (const float*)d_in[2];
    const float* ln1_g = (const float*)d_in[3];
    const float* ln1_b = (const float*)d_in[4];
    const float* ln2_g = (const float*)d_in[5];
    const float* ln2_b = (const float*)d_in[6];
    // per-name params: 0 mu_w, 1 proto_w, 2 mu_b, 3 gate_w, 4 pt_w, 5 pln_g, 6 pln_b, 7 prev
    auto W = [&](int n, int j) { return (const float*)d_in[7 + n * 8 + j]; };

    const size_t HH  = (size_t)TH * TH;
    const size_t BSH = (size_t)TNT * TH;
    const size_t BSS = (size_t)TB * TS * TS;
    const float  rsH = 1.0f / sqrtf((float)TH);

    // ---- f32 workspace
    float* ws     = (float*)d_ws;
    float* cmp    = ws;                 // HH
    float* mtc    = cmp + HH;
    float* gte    = mtc + HH;
    float* ln1    = gte + HH;           // BSH; recycled as f32 roped-q sink
    float* ln2    = ln1 + BSH;          // BSH; recycled as attn_out f32
    float* scores = ln2 + BSH;          // BSS
    float* x1     = scores + BSS;       // BSH
    float* hactf  = x1 + BSH;           // BSH
    // ---- f16 workspace
    _Float16* h16    = (_Float16*)(hactf + BSH);
    _Float16* eff16  = h16;             // 6*HH
    _Float16* wa16   = eff16 + 6 * HH;  // HH (prev / mu_w)
    _Float16* wb16   = wa16 + HH;       // HH (pt_w / gate_w)
    _Float16* ln1_16 = wb16 + HH;       // BSH
    _Float16* ln2_16 = ln1_16 + BSH;    // BSH
    _Float16* q16    = ln2_16 + BSH;    // BSH
    _Float16* k16    = q16 + BSH;       // BSH
    _Float16* v16    = k16 + BSH;       // BSH
    _Float16* attn16 = v16 + BSH;       // BSH
    _Float16* hact16 = attn16 + BSH;    // BSH
    _Float16* pr16   = hact16 + BSH;    // BSS (probs)

    float* out      = (float*)d_out;
    float* o_x      = out;
    float* o_masked = out + BSH;        // 6 slices
    float* o_raw    = out + 7  * BSH;
    float* o_logits = out + 13 * BSH;
    float* o_pk     = out + 19 * BSH;
    float* o_pv     = out + 20 * BSH;

    const dim3 blk(256);
    const dim3 gWW(TH / BN, TH / BM, 1);
    const unsigned ebl = (unsigned)((BSH + 255) / 256);

    auto cvt = [&](const float* src, _Float16* dst, size_t n) {
        cvt16_kernel<<<dim3((unsigned)((n / 4 + 255) / 256)), blk, 0, stream>>>(src, dst, (long)n);
    };

    // ---- Phase 1: eff16[n] = f16(proto + LN(prev @ pt.T))
    for (int n = 0; n < 6; ++n) {
        cvt(W(n, 7), wa16, HH);
        cvt(W(n, 4), wb16, HH);
        gemm_wmma<true><<<gWW, blk, 0, stream>>>(wa16, wb16, nullptr, cmp, nullptr,
                                                 TH, TH, TH, 0, 0, 0, 1.0f);
        ln_kernel<<<dim3(TH), blk, 0, stream>>>(cmp, W(n, 5), W(n, 6), W(n, 1),
                                                nullptr, eff16 + n * HH);
    }

    // ---- Phase 2: ln1 / ln2 (f32 residual + f16 GEMM operand)
    ln_kernel<<<dim3(TNT), blk, 0, stream>>>(x, ln1_g, ln1_b, nullptr, ln1, ln1_16);
    ln_kernel<<<dim3(TNT), blk, 0, stream>>>(x, ln2_g, ln2_b, nullptr, ln2, ln2_16);

    // ---- Phase 3: q/k/v split-route (names 0,1,2)
    for (int n = 0; n < 3; ++n) {
        cvt(W(n, 0), wa16, HH);
        cvt(W(n, 3), wb16, HH);
        gemm_wmma<true><<<gWW, blk, 0, stream>>>(ln1_16, wa16, W(n, 2), cmp, nullptr, TNT, TH, TH, 0, 0, 0, 1.0f);
        gemm_wmma<true><<<gWW, blk, 0, stream>>>(ln1_16, eff16 + n * HH, nullptr, mtc, nullptr, TNT, TH, TH, 0, 0, 0, rsH);
        gemm_wmma<true><<<gWW, blk, 0, stream>>>(ln1_16, wb16, nullptr, gte, nullptr, TNT, TH, TH, 0, 0, 0, 1.0f);
        route_kernel<<<dim3(TNT), blk, 0, stream>>>(cmp, mtc, gte, ln1,
            o_masked + n * BSH, o_raw + n * BSH, o_logits + n * BSH);
    }

    // ---- Phase 4: f1 split-route (name 4), residual = ln2
    cvt(W(4, 0), wa16, HH);
    cvt(W(4, 3), wb16, HH);
    gemm_wmma<true><<<gWW, blk, 0, stream>>>(ln2_16, wa16, W(4, 2), cmp, nullptr, TNT, TH, TH, 0, 0, 0, 1.0f);
    gemm_wmma<true><<<gWW, blk, 0, stream>>>(ln2_16, eff16 + 4 * HH, nullptr, mtc, nullptr, TNT, TH, TH, 0, 0, 0, rsH);
    gemm_wmma<true><<<gWW, blk, 0, stream>>>(ln2_16, wb16, nullptr, gte, nullptr, TNT, TH, TH, 0, 0, 0, 1.0f);
    route_kernel<<<dim3(TNT), blk, 0, stream>>>(cmp, mtc, gte, ln2,
        o_masked + 4 * BSH, o_raw + 4 * BSH, o_logits + 4 * BSH);

    // ---- Phase 5: RoPE + present kv
    rope_kernel<<<dim3(ebl), blk, 0, stream>>>(o_masked + 0 * BSH, cosb, sinb, ln1,  q16, (long)BSH);
    rope_kernel<<<dim3(ebl), blk, 0, stream>>>(o_masked + 1 * BSH, cosb, sinb, o_pk, k16, (long)BSH);
    copy_kernel<<<dim3(ebl), blk, 0, stream>>>(o_masked + 2 * BSH, o_pv, (long)BSH);
    cvt(o_masked + 2 * BSH, v16, BSH);

    // ---- Phase 6: attention
    gemm_wmma<true><<<dim3(TS / BN, TS / BM, TB), blk, 0, stream>>>(
        q16, k16, nullptr, scores, nullptr, TS, TS, TH,
        (long)TS * TH, (long)TS * TH, (long)TS * TS, rsH);
    softmax_kernel<<<dim3(TB * TS), blk, 0, stream>>>(scores, pr16);
    gemm_wmma<false><<<dim3(TH / BN, TS / BM, TB), blk, 0, stream>>>(
        pr16, v16, nullptr, ln2, attn16, TS, TH, TS,
        (long)TS * TS, (long)TS * TH, (long)TS * TH, 1.0f);

    // ---- Phase 7: o split-route (name 3), residual = attn_out(ln2 slot)
    cvt(W(3, 0), wa16, HH);
    cvt(W(3, 3), wb16, HH);
    gemm_wmma<true><<<gWW, blk, 0, stream>>>(attn16, wa16, W(3, 2), cmp, nullptr, TNT, TH, TH, 0, 0, 0, 1.0f);
    gemm_wmma<true><<<gWW, blk, 0, stream>>>(attn16, eff16 + 3 * HH, nullptr, mtc, nullptr, TNT, TH, TH, 0, 0, 0, rsH);
    gemm_wmma<true><<<gWW, blk, 0, stream>>>(attn16, wb16, nullptr, gte, nullptr, TNT, TH, TH, 0, 0, 0, 1.0f);
    route_kernel<<<dim3(TNT), blk, 0, stream>>>(cmp, mtc, gte, ln2,
        o_masked + 3 * BSH, o_raw + 3 * BSH, o_logits + 3 * BSH);
    add_kernel<<<dim3(ebl), blk, 0, stream>>>(x, o_masked + 3 * BSH, x1, (long)BSH);

    // ---- Phase 8: f2 path on relu(m1)
    relu_kernel<<<dim3(ebl), blk, 0, stream>>>(o_masked + 4 * BSH, hactf, hact16, (long)BSH);
    cvt(W(5, 0), wa16, HH);
    cvt(W(5, 3), wb16, HH);
    gemm_wmma<true><<<gWW, blk, 0, stream>>>(hact16, wa16, W(5, 2), cmp, nullptr, TNT, TH, TH, 0, 0, 0, 1.0f);
    gemm_wmma<true><<<gWW, blk, 0, stream>>>(hact16, eff16 + 5 * HH, nullptr, mtc, nullptr, TNT, TH, TH, 0, 0, 0, rsH);
    gemm_wmma<true><<<gWW, blk, 0, stream>>>(hact16, wb16, nullptr, gte, nullptr, TNT, TH, TH, 0, 0, 0, 1.0f);
    route_kernel<<<dim3(TNT), blk, 0, stream>>>(cmp, mtc, gte, hactf,
        o_masked + 5 * BSH, o_raw + 5 * BSH, o_logits + 5 * BSH);
    add_kernel<<<dim3(ebl), blk, 0, stream>>>(x1, o_masked + 5 * BSH, o_x, (long)BSH);
}
